// ALIGNNConv_18519898980955
// MI455X (gfx1250) — matrix operand order repaired and checked by reading, not compiled, not run on hardware
//
#include <hip/hip_runtime.h>

// ---------------------------------------------------------------------------
// Types for CDNA5 WMMA (wave32): D(16x16 f32) = A(16x32 bf16) x B(32x16 bf16) + C
// ---------------------------------------------------------------------------
typedef __attribute__((ext_vector_type(16))) __bf16 v16bf;
typedef __attribute__((ext_vector_type(8)))  __bf16 v8bf;
typedef __attribute__((ext_vector_type(4)))  __bf16 v4bf;
typedef __attribute__((ext_vector_type(8)))  float  v8f;

__device__ __forceinline__ __bf16 f2bf(float f) {
  unsigned u = __builtin_bit_cast(unsigned, f);
  unsigned r = (u + 0x7FFFu + ((u >> 16) & 1u)) >> 16;
  return __builtin_bit_cast(__bf16, (unsigned short)r);
}

__device__ __forceinline__ float silu_aff(float x, float a, float c) {
  float s = a * x + c;
  return s / (1.0f + __expf(-s));
}

__device__ __forceinline__ float sigmoidf_(float x) {
  return 1.0f / (1.0f + __expf(-x));
}

// ---------------------------------------------------------------------------
// Fused GEMM:  H[R,NO] = act(X[R,K]) @ W[NO,K]^T + bias
//   act(x) = (actA ? silu(actA[k]*x + actC[k]) : x)   (BN+SiLU of producer)
//   optional per-column sum / sum-of-squares accumulation for the next BN.
// 8 waves/block, 128 rows/block. bf16 operands staged in LDS (float4-in,
// ds_store_b64-out), f32 accumulators.
// ---------------------------------------------------------------------------
template <int K, int NO>
__global__ __launch_bounds__(256) void gemm_bn_kern(
    int R, const float* __restrict__ X,
    const float* __restrict__ actA, const float* __restrict__ actC,
    const float* __restrict__ W, const float* __restrict__ bias,
    float* __restrict__ H, float* __restrict__ sSum, float* __restrict__ sSq) {
  constexpr int ROWS = 128;     // rows per block
  constexpr int AST  = K + 16;  // padded LDS row stride (bf16 elements)
  constexpr int NKT  = K / 32;  // K tiles of 32
  constexpr int NT   = NO / 16; // N tiles of 16
  constexpr int KV   = K / 4;   // float4 chunks per row

  __shared__ __align__(16) __bf16 sA[ROWS * AST];
  __shared__ __align__(16) __bf16 sW[NO * AST];

  const int tid     = threadIdx.x;
  const int rowbase = blockIdx.x * ROWS;
  const bool full   = (rowbase + ROWS) <= R;

  // stage weights (row-major [NO,K]) as bf16, float4 granularity
  for (int i = tid; i < NO * KV; i += 256) {
    int n = i / KV, k4 = i - n * KV;
    float4 v = *(const float4*)(W + (size_t)n * K + k4 * 4);
    v4bf p;
    p[0] = f2bf(v.x); p[1] = f2bf(v.y); p[2] = f2bf(v.z); p[3] = f2bf(v.w);
    *(v4bf*)(sA ? sW + n * AST + k4 * 4 : nullptr) = p;  // (sA test folds away)
  }
  // stage activations; apply producer BN+SiLU on the fly; zero-pad tail rows
  for (int i = tid; i < ROWS * KV; i += 256) {
    int m = i / KV, k4 = i - m * KV;
    int k = k4 * 4;
    float4 v = {0.0f, 0.0f, 0.0f, 0.0f};
    if (full || (rowbase + m) < R)
      v = *(const float4*)(X + (size_t)(rowbase + m) * K + k);
    if (actA) {
      float4 av = *(const float4*)(actA + k);
      float4 cv = *(const float4*)(actC + k);
      v.x = silu_aff(v.x, av.x, cv.x);
      v.y = silu_aff(v.y, av.y, cv.y);
      v.z = silu_aff(v.z, av.z, cv.z);
      v.w = silu_aff(v.w, av.w, cv.w);
    }
    v4bf p;
    p[0] = f2bf(v.x); p[1] = f2bf(v.y); p[2] = f2bf(v.z); p[3] = f2bf(v.w);
    *(v4bf*)(sA + m * AST + k) = p;
  }
  __syncthreads();

  const int lane   = tid & 31;
  const int wave   = tid >> 5;
  const int m16    = lane & 15;
  const int hi4    = lane >> 4;
  const int rowoff = wave * 16;

  // A fragments: lane half hi4 covers K = kt*32 + hi4*8 + {0..7} and +16..23
  v16bf afrag[NKT];
#pragma unroll
  for (int kt = 0; kt < NKT; ++kt) {
    const __bf16* p = sA + (rowoff + m16) * AST + kt * 32 + hi4 * 8;
    v8bf lo = *(const v8bf*)p;
    v8bf hi = *(const v8bf*)(p + 16);
    afrag[kt] = __builtin_shufflevector(lo, hi, 0, 1, 2, 3, 4, 5, 6, 7, 8, 9,
                                        10, 11, 12, 13, 14, 15);
  }

#pragma unroll
  for (int nt = 0; nt < NT; ++nt) {
    const int n = nt * 16 + m16;
    float bn = bias ? bias[n] : 0.0f;
    v8f acc = {bn, bn, bn, bn, bn, bn, bn, bn};
#pragma unroll
    for (int kt = 0; kt < NKT; ++kt) {
      // B fragment: lane = output column n; half hi4 covers K = kt*32+hi4*8+{0..7} and +16..23
      const __bf16* p = sW + n * AST + kt * 32 + hi4 * 8;
      v8bf lo = *(const v8bf*)p;
      v8bf hi = *(const v8bf*)(p + 16);
      v16bf bfrag = __builtin_shufflevector(lo, hi, 0, 1, 2, 3, 4, 5, 6, 7, 8,
                                            9, 10, 11, 12, 13, 14, 15);
      acc = __builtin_amdgcn_wmma_f32_16x16x32_bf16(
          false, afrag[kt], false, bfrag, (short)0, acc, false, false);
    }
    const int rb = rowbase + rowoff + hi4 * 8;
    float* __restrict__ hp = H + (size_t)rb * NO + n;
    if (full) {
      // contiguous-immediate stores, no per-element guards
#pragma unroll
      for (int v = 0; v < 8; ++v) hp[v * NO] = acc[v];
      if (sSum) {
        float s = 0.0f, q = 0.0f;
#pragma unroll
        for (int v = 0; v < 8; ++v) { s += acc[v]; q += acc[v] * acc[v]; }
        s += __shfl_xor(s, 16, 32);
        q += __shfl_xor(q, 16, 32);
        if (hi4 == 0) {
          atomicAdd(&sSum[n], s);
          atomicAdd(&sSq[n], q);
        }
      }
    } else {
#pragma unroll
      for (int v = 0; v < 8; ++v) {
        if (rb + v < R) hp[v * NO] = acc[v];
      }
      if (sSum) {
        float s = 0.0f, q = 0.0f;
#pragma unroll
        for (int v = 0; v < 8; ++v) {
          if (rb + v < R) { s += acc[v]; q += acc[v] * acc[v]; }
        }
        s += __shfl_xor(s, 16, 32);
        q += __shfl_xor(q, 16, 32);
        if (hi4 == 0) {
          atomicAdd(&sSum[n], s);
          atomicAdd(&sSq[n], q);
        }
      }
    }
  }
}

// BN finalize: a = g * rsqrt(var + eps), c = be - mu * a
__global__ void finalize_kern(const float* __restrict__ sum,
                              const float* __restrict__ sq,
                              const float* __restrict__ g,
                              const float* __restrict__ be, float invR, int NO,
                              float* __restrict__ a, float* __restrict__ c) {
  int i = threadIdx.x;
  if (i < NO) {
    float mu  = sum[i] * invR;
    float var = sq[i] * invR - mu * mu;
    float rs  = rsqrtf(var + 1e-5f);
    float ai  = g[i] * rs;
    a[i] = ai;
    c[i] = be[i] - mu * ai;
  }
}

// concat 4 x [32,32] gate weights -> [128,32] (src_gate|dst_gate|dst_update|src_update)
__global__ void concat_gates_kern(const float* __restrict__ a,
                                  const float* __restrict__ b,
                                  const float* __restrict__ c,
                                  const float* __restrict__ d,
                                  float* __restrict__ out) {
  int i = blockIdx.x * 256 + threadIdx.x;
  if (i < 4096) {
    const float* s = (i < 1024) ? a : (i < 2048) ? b : (i < 3072) ? c : d;
    out[i] = s[i & 1023];
  }
}

// 4-lane float LDS tree reduction helper: red holds 4 partials per thread,
// all reduction offsets are multiples of 8 so column groups stay aligned.
// Edge scatter: m = Gh[s,0:32] + Gh[d,32:64] + M ; sigma = sigmoid(m)
// num[d] += Gh[s,64:96]*sigma ; den[d] += sigma ; M <- m ; stats(m)
// One thread = one edge x 4 columns.
__global__ __launch_bounds__(256) void eggc_scatter_kern(
    int Re, const int* __restrict__ esrc, const int* __restrict__ edst,
    const float* __restrict__ Gh, float* __restrict__ M,
    float* __restrict__ num, float* __restrict__ den,
    float* __restrict__ sSum, float* __restrict__ sSq) {
  __shared__ float red[256 * 4];
  const int tid = threadIdx.x;
  const int i   = blockIdx.x * 256 + tid;
  const int cg  = (tid & 7) * 4;  // column group (0,4,...,28)
  float ls0 = 0, ls1 = 0, ls2 = 0, ls3 = 0;
  float lq0 = 0, lq1 = 0, lq2 = 0, lq3 = 0;
  if (i < Re * 8) {
    int t = i >> 3;
    int s = esrc[t], d = edst[t];
    float4 gs = *(const float4*)(Gh + (size_t)s * 128 + cg);
    float4 gd = *(const float4*)(Gh + (size_t)d * 128 + 32 + cg);
    float4 gu = *(const float4*)(Gh + (size_t)s * 128 + 64 + cg);
    float4 mv = *(const float4*)(M + (size_t)t * 32 + cg);
    float m0 = gs.x + gd.x + mv.x, m1 = gs.y + gd.y + mv.y;
    float m2 = gs.z + gd.z + mv.z, m3 = gs.w + gd.w + mv.w;
    float g0 = sigmoidf_(m0), g1 = sigmoidf_(m1);
    float g2 = sigmoidf_(m2), g3 = sigmoidf_(m3);
    float* np = num + (size_t)d * 32 + cg;
    float* dp = den + (size_t)d * 32 + cg;
    atomicAdd(np + 0, gu.x * g0);
    atomicAdd(np + 1, gu.y * g1);
    atomicAdd(np + 2, gu.z * g2);
    atomicAdd(np + 3, gu.w * g3);
    atomicAdd(dp + 0, g0);
    atomicAdd(dp + 1, g1);
    atomicAdd(dp + 2, g2);
    atomicAdd(dp + 3, g3);
    float4 mo = {m0, m1, m2, m3};
    *(float4*)(M + (size_t)t * 32 + cg) = mo;
    ls0 = m0; ls1 = m1; ls2 = m2; ls3 = m3;
    lq0 = m0 * m0; lq1 = m1 * m1; lq2 = m2 * m2; lq3 = m3 * m3;
  }
  red[tid * 4 + 0] = ls0; red[tid * 4 + 1] = ls1;
  red[tid * 4 + 2] = ls2; red[tid * 4 + 3] = ls3;
  __syncthreads();
#pragma unroll
  for (int off = 128; off >= 8; off >>= 1) {
    if (tid < off) {
#pragma unroll
      for (int j = 0; j < 4; ++j) red[tid * 4 + j] += red[(tid + off) * 4 + j];
    }
    __syncthreads();
  }
  if (tid < 8) {
#pragma unroll
    for (int j = 0; j < 4; ++j) atomicAdd(&sSum[tid * 4 + j], red[tid * 4 + j]);
  }
  __syncthreads();
  red[tid * 4 + 0] = lq0; red[tid * 4 + 1] = lq1;
  red[tid * 4 + 2] = lq2; red[tid * 4 + 3] = lq3;
  __syncthreads();
#pragma unroll
  for (int off = 128; off >= 8; off >>= 1) {
    if (tid < off) {
#pragma unroll
      for (int j = 0; j < 4; ++j) red[tid * 4 + j] += red[(tid + off) * 4 + j];
    }
    __syncthreads();
  }
  if (tid < 8) {
#pragma unroll
    for (int j = 0; j < 4; ++j) atomicAdd(&sSq[tid * 4 + j], red[tid * 4 + j]);
  }
}

// Node combine: hn = Gh[r,96:128] + num/(den+1e-6) ; stats(hn)
// One thread = one node x 4 columns.
__global__ __launch_bounds__(256) void eggc_combine_kern(
    int Rn, const float* __restrict__ Gh, const float* __restrict__ num,
    const float* __restrict__ den, float* __restrict__ HN,
    float* __restrict__ sSum, float* __restrict__ sSq) {
  __shared__ float red[256 * 4];
  const int tid = threadIdx.x;
  const int i   = blockIdx.x * 256 + tid;
  const int cg  = (tid & 7) * 4;
  float ls0 = 0, ls1 = 0, ls2 = 0, ls3 = 0;
  float lq0 = 0, lq1 = 0, lq2 = 0, lq3 = 0;
  if (i < Rn * 8) {
    int r = i >> 3;
    float4 gu = *(const float4*)(Gh + (size_t)r * 128 + 96 + cg);
    float4 nv = *(const float4*)(num + (size_t)r * 32 + cg);
    float4 dv = *(const float4*)(den + (size_t)r * 32 + cg);
    float h0 = gu.x + nv.x / (dv.x + 1e-6f);
    float h1 = gu.y + nv.y / (dv.y + 1e-6f);
    float h2 = gu.z + nv.z / (dv.z + 1e-6f);
    float h3 = gu.w + nv.w / (dv.w + 1e-6f);
    float4 ho = {h0, h1, h2, h3};
    *(float4*)(HN + (size_t)r * 32 + cg) = ho;
    ls0 = h0; ls1 = h1; ls2 = h2; ls3 = h3;
    lq0 = h0 * h0; lq1 = h1 * h1; lq2 = h2 * h2; lq3 = h3 * h3;
  }
  red[tid * 4 + 0] = ls0; red[tid * 4 + 1] = ls1;
  red[tid * 4 + 2] = ls2; red[tid * 4 + 3] = ls3;
  __syncthreads();
#pragma unroll
  for (int off = 128; off >= 8; off >>= 1) {
    if (tid < off) {
#pragma unroll
      for (int j = 0; j < 4; ++j) red[tid * 4 + j] += red[(tid + off) * 4 + j];
    }
    __syncthreads();
  }
  if (tid < 8) {
#pragma unroll
    for (int j = 0; j < 4; ++j) atomicAdd(&sSum[tid * 4 + j], red[tid * 4 + j]);
  }
  __syncthreads();
  red[tid * 4 + 0] = lq0; red[tid * 4 + 1] = lq1;
  red[tid * 4 + 2] = lq2; red[tid * 4 + 3] = lq3;
  __syncthreads();
#pragma unroll
  for (int off = 128; off >= 8; off >>= 1) {
    if (tid < off) {
#pragma unroll
      for (int j = 0; j < 4; ++j) red[tid * 4 + j] += red[(tid + off) * 4 + j];
    }
    __syncthreads();
  }
  if (tid < 8) {
#pragma unroll
    for (int j = 0; j < 4; ++j) atomicAdd(&sSq[tid * 4 + j], red[tid * 4 + j]);
  }
}

// out = silu(a*H + c) + resid    (feature dim 128), float4-vectorized
__global__ void bn_silu_resid_kern(int n4, const float* __restrict__ Hm,
                                   const float* __restrict__ a,
                                   const float* __restrict__ c,
                                   const float* __restrict__ resid,
                                   float* __restrict__ out) {
  int i = blockIdx.x * 256 + threadIdx.x;
  if (i < n4) {
    int cc = (i & 31) * 4;
    float4 h  = *(const float4*)(Hm + (size_t)i * 4);
    float4 rv = *(const float4*)(resid + (size_t)i * 4);
    float4 av = *(const float4*)(a + cc);
    float4 cv = *(const float4*)(c + cc);
    float4 o;
    o.x = silu_aff(h.x, av.x, cv.x) + rv.x;
    o.y = silu_aff(h.y, av.y, cv.y) + rv.y;
    o.z = silu_aff(h.z, av.z, cv.z) + rv.z;
    o.w = silu_aff(h.w, av.w, cv.w) + rv.w;
    *(float4*)(out + (size_t)i * 4) = o;
  }
}

// ---------------------------------------------------------------------------
// Host-side orchestration
// ---------------------------------------------------------------------------
#define LAUNCH_GEMM(Kc, NOc, R, X, ACT, W, B, H, ST)                           \
  gemm_bn_kern<Kc, NOc><<<((R) + 127) / 128, 256, 0, stream>>>(                \
      (R), (X), (ACT) ? (ACT) + 256 : nullptr, (ACT) ? (ACT) + 384 : nullptr,  \
      (W), (B), (H), (ST), (ST) ? (ST) + 128 : nullptr)

#define LAUNCH_FIN(SL, NOc, G, BE, R)                                          \
  finalize_kern<<<1, (NOc), 0, stream>>>((SL), (SL) + 128, (G), (BE),          \
                                         1.0f / (float)(R), (NOc),             \
                                         (SL) + 256, (SL) + 384)

#define LAUNCH_APPLY(R, H, SL, RES, OUT)                                       \
  bn_silu_resid_kern<<<((R) * 32 + 255) / 256, 256, 0, stream>>>(              \
      (R) * 32, (H), (SL) + 256, (SL) + 384, (RES), (OUT))

extern "C" void kernel_launch(void* const* d_in, const int* in_sizes, int n_in,
                              void* d_out, int out_size, void* d_ws,
                              size_t ws_size, hipStream_t stream) {
  (void)in_sizes; (void)n_in; (void)out_size; (void)ws_size;
  const int N = 10000, E = 160000, T = 640000;

  const float* x = (const float*)d_in[0];
  const float* y = (const float*)d_in[1];
  const float* z = (const float*)d_in[2];
  const int* src  = (const int*)d_in[3];
  const int* dst  = (const int*)d_in[4];
  const int* lsrc = (const int*)d_in[5];
  const int* ldst = (const int*)d_in[6];

  // params flattened in jax pytree (sorted-key) order
  auto F = [&](int i) { return (const float*)d_in[i]; };
  // edge_upd: bnb_e,bnb_n,bng_e,bng_n,dst_gate,dst_update,edge_gate,src_gate,src_update
  const float *eu_bnb_e = F(7), *eu_bnb_n = F(8), *eu_bng_e = F(9), *eu_bng_n = F(10);
  const float *eu_dgate = F(11), *eu_dupd = F(12), *eu_egate = F(13), *eu_sgate = F(14), *eu_supd = F(15);
  // node_bn: W1,W2,b1,b2,be1,be2,g1,g2
  const float *nb_W1 = F(16), *nb_W2 = F(17), *nb_b1 = F(18), *nb_b2 = F(19);
  const float *nb_be1 = F(20), *nb_be2 = F(21), *nb_g1 = F(22), *nb_g2 = F(23);
  // node_ex: W,b,be,g
  const float *nex_W = F(24), *nex_b = F(25), *nex_be = F(26), *nex_g = F(27);
  // node_upd
  const float *nu_bnb_e = F(28), *nu_bnb_n = F(29), *nu_bng_e = F(30), *nu_bng_n = F(31);
  const float *nu_dgate = F(32), *nu_dupd = F(33), *nu_egate = F(34), *nu_sgate = F(35), *nu_supd = F(36);
  // pair_bn
  const float *pb_W1 = F(37), *pb_W2 = F(38), *pb_b1 = F(39), *pb_b2 = F(40);
  const float *pb_be1 = F(41), *pb_be2 = F(42), *pb_g1 = F(43), *pb_g2 = F(44);
  // pair_ex
  const float *pex_W = F(45), *pex_b = F(46), *pex_be = F(47), *pex_g = F(48);
  // trip_bn
  const float *tb_W1 = F(49), *tb_W2 = F(50), *tb_b1 = F(51), *tb_b2 = F(52);
  const float *tb_be1 = F(53), *tb_be2 = F(54), *tb_g1 = F(55), *tb_g2 = F(56);
  // trip_ex
  const float *tex_W = F(57), *tex_b = F(58), *tex_be = F(59), *tex_g = F(60);

  // ---- workspace layout (floats), lifetime-based reuse ----
  float* ws = (float*)d_ws;
  size_t off = 0;
  auto alloc = [&](size_t n) { float* p = ws + off; off += n; return p; };
  float* R0   = alloc((size_t)T * 128);       // stage1 scratch (front T*64) then HZ
  float* B1   = R0;                           // bottleneck stage-1 output (<= T*64)
  float* Z32  = R0 + (size_t)T * 64;          // z bottleneck output [T,32]
  float* X32  = alloc((size_t)N * 32);
  float* Y32  = alloc((size_t)E * 32);
  float* R1   = alloc((size_t)E * 128);       // GhY, later HY
  float* MBUF = alloc((size_t)T * 32);        // edge_gate(z) then m (edge_upd)
  float* NUME = alloc((size_t)E * 32);
  float* DENE = alloc((size_t)E * 32);
  float* HNE  = alloc((size_t)E * 32);        // hn of edge_upd (raw)
  float* R2   = alloc((size_t)N * 128);       // GhX, later HX
  float* M2   = alloc((size_t)E * 32);        // edge_gate(m) then m (node_upd)
  float* NUMN = alloc((size_t)N * 32);
  float* DENN = alloc((size_t)N * 32);
  float* HNN  = alloc((size_t)N * 32);        // hn of node_upd (raw)
  float* WGE  = alloc(4096);
  float* WGN  = alloc(4096);
  float* STATS = alloc(13 * 512);             // 13 slots: sum/sq/a/c (128 each)
  auto S = [&](int j) { return STATS + (size_t)j * 512; };

  hipMemsetAsync(NUME, 0, (size_t)2 * E * 32 * sizeof(float), stream);  // NUME+DENE
  hipMemsetAsync(NUMN, 0, (size_t)2 * N * 32 * sizeof(float), stream);  // NUMN+DENN
  hipMemsetAsync(STATS, 0, (size_t)13 * 512 * sizeof(float), stream);

  float* out_x = (float*)d_out;
  float* out_y = out_x + (size_t)N * 128;
  float* out_z = out_y + (size_t)E * 128;

  // ---- bottlenecks (Linear->BN->SiLU twice); BN+SiLU fused into next GEMM load
  LAUNCH_GEMM(128, 64, N, x, (float*)nullptr, nb_W1, nb_b1, B1, S(0));
  LAUNCH_FIN(S(0), 64, nb_g1, nb_be1, N);
  LAUNCH_GEMM(64, 32, N, B1, S(0), nb_W2, nb_b2, X32, S(1));
  LAUNCH_FIN(S(1), 32, nb_g2, nb_be2, N);

  LAUNCH_GEMM(128, 64, E, y, (float*)nullptr, pb_W1, pb_b1, B1, S(2));
  LAUNCH_FIN(S(2), 64, pb_g1, pb_be1, E);
  LAUNCH_GEMM(64, 32, E, B1, S(2), pb_W2, pb_b2, Y32, S(3));
  LAUNCH_FIN(S(3), 32, pb_g2, pb_be2, E);

  LAUNCH_GEMM(128, 64, T, z, (float*)nullptr, tb_W1, tb_b1, B1, S(4));
  LAUNCH_FIN(S(4), 64, tb_g1, tb_be1, T);
  LAUNCH_GEMM(64, 32, T, B1, S(4), tb_W2, tb_b2, Z32, S(5));
  LAUNCH_FIN(S(5), 32, tb_g2, tb_be2, T);

  // ---- edge_upd EGGC: h = y32(act S3), e = z32(act S5), edges lsrc/ldst over T
  concat_gates_kern<<<16, 256, 0, stream>>>(eu_sgate, eu_dgate, eu_dupd, eu_supd, WGE);
  LAUNCH_GEMM(32, 128, E, Y32, S(3), WGE, (float*)nullptr, R1, (float*)nullptr);
  LAUNCH_GEMM(32, 32, T, Z32, S(5), eu_egate, (float*)nullptr, MBUF, (float*)nullptr);
  eggc_scatter_kern<<<(T * 8 + 255) / 256, 256, 0, stream>>>(
      T, lsrc, ldst, R1, MBUF, NUME, DENE, S(6), S(6) + 128);
  LAUNCH_FIN(S(6), 32, eu_bng_e, eu_bnb_e, T);
  eggc_combine_kern<<<(E * 8 + 255) / 256, 256, 0, stream>>>(
      E, R1, NUME, DENE, HNE, S(7), S(7) + 128);
  LAUNCH_FIN(S(7), 32, eu_bng_n, eu_bnb_n, E);

  // ---- node_upd EGGC: h = x32(act S1), e = m(HNE, act S7), edges src/dst over E
  concat_gates_kern<<<16, 256, 0, stream>>>(nu_sgate, nu_dgate, nu_dupd, nu_supd, WGN);
  LAUNCH_GEMM(32, 128, N, X32, S(1), WGN, (float*)nullptr, R2, (float*)nullptr);
  LAUNCH_GEMM(32, 32, E, HNE, S(7), nu_egate, (float*)nullptr, M2, (float*)nullptr);
  eggc_scatter_kern<<<(E * 8 + 255) / 256, 256, 0, stream>>>(
      E, src, dst, R2, M2, NUMN, DENN, S(8), S(8) + 128);
  LAUNCH_FIN(S(8), 32, nu_bng_e, nu_bnb_e, E);
  eggc_combine_kern<<<(N * 8 + 255) / 256, 256, 0, stream>>>(
      N, R2, NUMN, DENN, HNN, S(9), S(9) + 128);
  LAUNCH_FIN(S(9), 32, nu_bng_n, nu_bnb_n, N);

  // ---- expansion MLPs (32->128, BN+SiLU, +residual)
  LAUNCH_GEMM(32, 128, N, HNN, S(9), nex_W, nex_b, R2, S(10));
  LAUNCH_FIN(S(10), 128, nex_g, nex_be, N);
  LAUNCH_APPLY(N, R2, S(10), x, out_x);

  LAUNCH_GEMM(32, 128, E, M2, S(8), pex_W, pex_b, R1, S(11));
  LAUNCH_FIN(S(11), 128, pex_g, pex_be, E);
  LAUNCH_APPLY(E, R1, S(11), y, out_y);

  LAUNCH_GEMM(32, 128, T, MBUF, S(6), tex_W, tex_b, R0, S(12));
  LAUNCH_FIN(S(12), 128, tex_g, tex_be, T);
  LAUNCH_APPLY(T, R0, S(12), z, out_z);
}
